// MultiHeadAttention_42125039239485
// MI455X (gfx1250) — matrix operand order repaired and checked by reading, not compiled, not run on hardware
//
#include <hip/hip_runtime.h>

// ---------------------------------------------------------------------------
// MultiHeadAttention for MI455X (gfx1250): f16 WMMA with f32 accumulate.
// Stages: convert/transpose -> QKV GEMM -> flash attention -> output GEMM.
// WMMA stages software-pipelined with explicit ping-pong fragment buffers
// (no register copies), score chains start from inline-zero C.
// ---------------------------------------------------------------------------

typedef __attribute__((ext_vector_type(16))) _Float16 v16h;
typedef __attribute__((ext_vector_type(8)))  _Float16 v8h;
typedef __attribute__((ext_vector_type(8)))  float    v8f;

#define HIDDEN   768
#define HEADS    12
#define HEAD_DIM 64
#define SEQ      4096

// Q pre-scale: 1/sqrt(64) * log2(e) so softmax can use native exp2.
#define QSCALE   (0.125f * 1.44269504088896340736f)

// ---- WMMA fragment helpers (layouts per CDNA5 ISA §7.12.2) ------

// A fragment, 16x32 f16, row-major source with leading dim `ld` (halves).
static __device__ inline v16h load_a16(const _Float16* __restrict__ p, int ld) {
    int lane = threadIdx.x & 31;
    const _Float16* r = p + (lane & 15) * ld + ((lane >> 4) << 3);
    union { v16h v; v8h h[2]; } u;
    u.h[0] = *reinterpret_cast<const v8h*>(r);
    u.h[1] = *reinterpret_cast<const v8h*>(r + 16);
    return u.v;
}

// B fragment, 32x16 f16. Source: one output-column per row (K contiguous).
static __device__ inline v16h load_b16(const _Float16* __restrict__ p, int ld) {
    int lane = threadIdx.x & 31;
    const _Float16* r = p + (lane & 15) * ld + ((lane >> 4) << 3);
    union { v16h v; v8h h[2]; } u;
    u.h[0] = *reinterpret_cast<const v8h*>(r);
    u.h[1] = *reinterpret_cast<const v8h*>(r + 16);
    return u.v;
}

static __device__ inline v8f wmma16(v16h a, v16h b, v8f c) {
    return __builtin_amdgcn_wmma_f32_16x16x32_f16(false, a, false, b,
                                                  (short)0, c, false, false);
}

// First WMMA of an accumulation chain: constant-zero C (SRC2 = inline 0).
static __device__ inline v8f wmma16z(v16h a, v16h b) {
    v8f z = {0.f, 0.f, 0.f, 0.f, 0.f, 0.f, 0.f, 0.f};
    return __builtin_amdgcn_wmma_f32_16x16x32_f16(false, a, false, b,
                                                  (short)0, z, false, false);
}

// ---- Stage 0: conversions / transposes -------------------------------------

__global__ __launch_bounds__(256) void cvt_f32_f16(const float* __restrict__ in,
                                                   _Float16* __restrict__ out, int n) {
    int i = blockIdx.x * 256 + threadIdx.x;
    if (i < n) out[i] = (_Float16)in[i];
}

// in: [H][D][dh] -> out: [H][dh][D]  (per-head transpose, f32 -> f16)
__global__ __launch_bounds__(256) void transpose_whead(const float* __restrict__ in,
                                                       _Float16* __restrict__ out) {
    int i = blockIdx.x * 256 + threadIdx.x;           // HEADS*HEAD_DIM*HIDDEN
    int h = i / (HEAD_DIM * HIDDEN);
    int rem = i - h * (HEAD_DIM * HIDDEN);
    int e = rem / HIDDEN;
    int d = rem - e * HIDDEN;
    out[i] = (_Float16)in[(h * HIDDEN + d) * HEAD_DIM + e];
}

// in: [D][D] -> out: [D][D] transposed (f32 -> f16)
__global__ __launch_bounds__(256) void transpose_wo(const float* __restrict__ in,
                                                    _Float16* __restrict__ out) {
    int i = blockIdx.x * 256 + threadIdx.x;           // HIDDEN*HIDDEN
    int j = i / HIDDEN;
    int d = i - j * HIDDEN;
    out[i] = (_Float16)in[d * HIDDEN + j];
}

// ---- Stage 1: QKV projection -----------------------------------------------
// One wave computes a 32x64 tile of Q, K, or V for one head.
// K-loop unrolled x2 with ping-pong fragment buffers (no copies).
__global__ __launch_bounds__(128) void qkv_kernel(
    const _Float16* __restrict__ xh,
    const _Float16* __restrict__ Wqt, const _Float16* __restrict__ Wkt,
    const _Float16* __restrict__ Wvt,
    const float* __restrict__ bq, const float* __restrict__ bk,
    const float* __restrict__ bv,
    _Float16* __restrict__ Qh, _Float16* __restrict__ Kh,
    _Float16* __restrict__ Vt) {

    int gid  = blockIdx.x * 4 + (threadIdx.x >> 5);
    int sel  = gid / (HEADS * 128);                   // 0=Q 1=K 2=V
    int rem  = gid - sel * (HEADS * 128);
    int h    = rem >> 7;
    int m0   = (rem & 127) << 5;                      // 32-row tile

    const _Float16* W    = (sel == 0) ? Wqt : (sel == 1) ? Wkt : Wvt;
    const float*    bias = (sel == 0) ? bq  : (sel == 1) ? bk  : bv;
    const _Float16* Wh   = W + h * (HEAD_DIM * HIDDEN);
    const _Float16* A0   = xh + m0 * HIDDEN;
    const _Float16* A1   = xh + (m0 + 16) * HIDDEN;

    v8f acc0[4] = {}, acc1[4] = {};

    // ping buffer: k-step 0
    v16h a0A = load_a16(A0, HIDDEN);
    v16h a1A = load_a16(A1, HIDDEN);
    v16h bA[4];
#pragma unroll
    for (int nt = 0; nt < 4; ++nt) bA[nt] = load_b16(Wh + nt * 16 * HIDDEN, HIDDEN);

    for (int kk = 0; kk < HIDDEN; kk += 64) {
        int k1 = kk + 32;                              // always < HIDDEN
        v16h a0B = load_a16(A0 + k1, HIDDEN);
        v16h a1B = load_a16(A1 + k1, HIDDEN);
        v16h bB[4];
#pragma unroll
        for (int nt = 0; nt < 4; ++nt)
            bB[nt] = load_b16(Wh + nt * 16 * HIDDEN + k1, HIDDEN);
#pragma unroll
        for (int nt = 0; nt < 4; ++nt) acc0[nt] = wmma16(a0A, bA[nt], acc0[nt]);
#pragma unroll
        for (int nt = 0; nt < 4; ++nt) acc1[nt] = wmma16(a1A, bA[nt], acc1[nt]);

        int k2 = (kk + 64 < HIDDEN) ? kk + 64 : 0;     // dead prefetch on last
        a0A = load_a16(A0 + k2, HIDDEN);
        a1A = load_a16(A1 + k2, HIDDEN);
#pragma unroll
        for (int nt = 0; nt < 4; ++nt)
            bA[nt] = load_b16(Wh + nt * 16 * HIDDEN + k2, HIDDEN);
#pragma unroll
        for (int nt = 0; nt < 4; ++nt) acc0[nt] = wmma16(a0B, bB[nt], acc0[nt]);
#pragma unroll
        for (int nt = 0; nt < 4; ++nt) acc1[nt] = wmma16(a1B, bB[nt], acc1[nt]);
    }

    int lane = threadIdx.x & 31;
    int col  = lane & 15;
    int rb   = (lane >> 4) << 3;                      // row base (0 or 8)
    float qscale = (sel == 0) ? QSCALE : 1.0f;

#pragma unroll
    for (int nt = 0; nt < 4; ++nt) {
        float bb = bias[h * HEAD_DIM + nt * 16 + col];
#pragma unroll
        for (int half = 0; half < 2; ++half) {
            const v8f& a = half ? acc1[nt] : acc0[nt];
#pragma unroll
            for (int r = 0; r < 8; ++r) {
                float v = (a[r] + bb) * qscale;
                int row = m0 + half * 16 + rb + r;
                int e   = nt * 16 + col;
                if (sel == 2)
                    Vt[(h * HEAD_DIM + e) * SEQ + row] = (_Float16)v;
                else if (sel == 0)
                    Qh[(h * SEQ + row) * HEAD_DIM + e] = (_Float16)v;
                else
                    Kh[(h * SEQ + row) * HEAD_DIM + e] = (_Float16)v;
            }
        }
    }
}

// ---- Stage 2: flash attention ----------------------------------------------
// grid (64, 12); block = 128 threads = 4 waves; wave owns 16 query rows.
__global__ __launch_bounds__(128) void attn_kernel(
    const _Float16* __restrict__ Qh, const _Float16* __restrict__ Kh,
    const _Float16* __restrict__ Vt, _Float16* __restrict__ ctx) {

    __shared__ _Float16 ldsP[4][16 * 64];             // per-wave P staging

    int wave = threadIdx.x >> 5;
    int lane = threadIdx.x & 31;
    int col  = lane & 15;
    int hw   = lane >> 4;                             // half-wave id
    int h    = blockIdx.y;
    int q0   = blockIdx.x * 64 + wave * 16;

    const _Float16* Qb = Qh + (h * SEQ + q0) * HEAD_DIM;
    const _Float16* Kb = Kh + h * SEQ * HEAD_DIM;
    const _Float16* Vb = Vt + h * HEAD_DIM * SEQ;
    _Float16* pP = &ldsP[wave][0];

    // Q fragments are loop-invariant: keep them resident.
    v16h aq0 = load_a16(Qb, HEAD_DIM);
    v16h aq1 = load_a16(Qb + 32, HEAD_DIM);

    v8f o[4] = {};
    float mrun[8], lrun[8], scale[8];
#pragma unroll
    for (int r = 0; r < 8; ++r) { mrun[r] = -3.0e38f; lrun[r] = 0.0f; }

    for (int kb = 0; kb < SEQ; kb += 64) {
        // ---- scores S = (Q*log2e/sqrt(dh)) @ K^T, pipelined over nt ------
        const _Float16* kp = Kb + kb * HEAD_DIM;
        v8f s[4];
        v16h bk0 = load_b16(kp, HEAD_DIM);
        v16h bk1 = load_b16(kp + 32, HEAD_DIM);
#pragma unroll
        for (int nt = 0; nt < 3; ++nt) {
            v16h n0 = load_b16(kp + (nt + 1) * 16 * HEAD_DIM, HEAD_DIM);
            v16h n1 = load_b16(kp + (nt + 1) * 16 * HEAD_DIM + 32, HEAD_DIM);
            s[nt] = wmma16z(aq0, bk0);                 // C = inline 0
            s[nt] = wmma16(aq1, bk1, s[nt]);
            bk0 = n0; bk1 = n1;                        // unrolled: SSA-renamed
        }
        s[3] = wmma16z(aq0, bk0);
        s[3] = wmma16(aq1, bk1, s[3]);

        // ---- prefetch V fragments (kk=0); softmax VALU hides the latency -
        v16h vb[4];
#pragma unroll
        for (int dt = 0; dt < 4; ++dt)
            vb[dt] = load_b16(Vb + dt * 16 * SEQ + kb, SEQ);

        // ---- online softmax (base-2; row stats via half-wave shfl) -------
#pragma unroll
        for (int r = 0; r < 8; ++r) {
            float m = fmaxf(fmaxf(s[0][r], s[1][r]), fmaxf(s[2][r], s[3][r]));
#pragma unroll
            for (int off = 1; off < 16; off <<= 1)
                m = fmaxf(m, __shfl_xor(m, off, 32));
            float mnew = fmaxf(mrun[r], m);
            float sc   = exp2f(mrun[r] - mnew);
            float ls   = 0.0f;
#pragma unroll
            for (int nt = 0; nt < 4; ++nt) {
                float p = exp2f(s[nt][r] - mnew);
                ls += p;
                pP[(hw * 8 + r) * 64 + nt * 16 + col] = (_Float16)p;
            }
#pragma unroll
            for (int off = 1; off < 16; off <<= 1)
                ls += __shfl_xor(ls, off, 32);
            lrun[r]  = lrun[r] * sc + ls;
            mrun[r]  = mnew;
            scale[r] = sc;
        }
#pragma unroll
        for (int dt = 0; dt < 4; ++dt)
#pragma unroll
            for (int r = 0; r < 8; ++r)
                o[dt][r] *= scale[r];

        // ---- O += P @ V, second half prefetched under first half ---------
        v16h ap0 = load_a16(pP, 64);
        v16h ap1 = load_a16(pP + 32, 64);
        v16h vb2[4];
#pragma unroll
        for (int dt = 0; dt < 4; ++dt)
            vb2[dt] = load_b16(Vb + dt * 16 * SEQ + kb + 32, SEQ);
#pragma unroll
        for (int dt = 0; dt < 4; ++dt) o[dt] = wmma16(ap0, vb[dt], o[dt]);
#pragma unroll
        for (int dt = 0; dt < 4; ++dt) o[dt] = wmma16(ap1, vb2[dt], o[dt]);
    }

    // ---- normalize, store ctx as f16 [S][H*dh] -------------------------
    float invl[8];
#pragma unroll
    for (int r = 0; r < 8; ++r) invl[r] = 1.0f / lrun[r];
#pragma unroll
    for (int dt = 0; dt < 4; ++dt)
#pragma unroll
        for (int r = 0; r < 8; ++r)
            ctx[(q0 + hw * 8 + r) * HIDDEN + h * HEAD_DIM + dt * 16 + col] =
                (_Float16)(o[dt][r] * invl[r]);
}

// ---- Stage 3: output projection ctx @ Wo + bo -> f32 ------------------------
// One wave per 32x64 output tile, ping-pong double-buffered K loop.
__global__ __launch_bounds__(128) void proj_kernel(
    const _Float16* __restrict__ ctx, const _Float16* __restrict__ Wot,
    const float* __restrict__ bo, float* __restrict__ out) {

    int gid = blockIdx.x * 4 + (threadIdx.x >> 5);
    int mt  = gid & 127;
    int nb  = gid >> 7;                               // 0..11
    int m0  = mt << 5;
    int n0  = nb * 64;

    const _Float16* A0 = ctx + m0 * HIDDEN;
    const _Float16* A1 = ctx + (m0 + 16) * HIDDEN;
    const _Float16* Wn = Wot + n0 * HIDDEN;

    v8f acc0[4] = {}, acc1[4] = {};
    v16h a0A = load_a16(A0, HIDDEN);
    v16h a1A = load_a16(A1, HIDDEN);
    v16h bA[4];
#pragma unroll
    for (int nt = 0; nt < 4; ++nt) bA[nt] = load_b16(Wn + nt * 16 * HIDDEN, HIDDEN);

    for (int kk = 0; kk < HIDDEN; kk += 64) {
        int k1 = kk + 32;
        v16h a0B = load_a16(A0 + k1, HIDDEN);
        v16h a1B = load_a16(A1 + k1, HIDDEN);
        v16h bB[4];
#pragma unroll
        for (int nt = 0; nt < 4; ++nt)
            bB[nt] = load_b16(Wn + nt * 16 * HIDDEN + k1, HIDDEN);
#pragma unroll
        for (int nt = 0; nt < 4; ++nt) acc0[nt] = wmma16(a0A, bA[nt], acc0[nt]);
#pragma unroll
        for (int nt = 0; nt < 4; ++nt) acc1[nt] = wmma16(a1A, bA[nt], acc1[nt]);

        int k2 = (kk + 64 < HIDDEN) ? kk + 64 : 0;
        a0A = load_a16(A0 + k2, HIDDEN);
        a1A = load_a16(A1 + k2, HIDDEN);
#pragma unroll
        for (int nt = 0; nt < 4; ++nt)
            bA[nt] = load_b16(Wn + nt * 16 * HIDDEN + k2, HIDDEN);
#pragma unroll
        for (int nt = 0; nt < 4; ++nt) acc0[nt] = wmma16(a0B, bB[nt], acc0[nt]);
#pragma unroll
        for (int nt = 0; nt < 4; ++nt) acc1[nt] = wmma16(a1B, bB[nt], acc1[nt]);
    }

    int lane = threadIdx.x & 31;
    int col  = lane & 15;
    int rb   = (lane >> 4) << 3;
#pragma unroll
    for (int nt = 0; nt < 4; ++nt) {
        float bb = bo[n0 + nt * 16 + col];
#pragma unroll
        for (int r = 0; r < 8; ++r) {
            out[(m0 + rb + r) * HIDDEN + n0 + nt * 16 + col] = acc0[nt][r] + bb;
            out[(m0 + 16 + rb + r) * HIDDEN + n0 + nt * 16 + col] = acc1[nt][r] + bb;
        }
    }
}

// ---------------------------------------------------------------------------
extern "C" void kernel_launch(void* const* d_in, const int* in_sizes, int n_in,
                              void* d_out, int out_size, void* d_ws, size_t ws_size,
                              hipStream_t stream) {
    const float* x  = (const float*)d_in[0];
    const float* Wq = (const float*)d_in[1];
    const float* Wk = (const float*)d_in[2];
    const float* Wv = (const float*)d_in[3];
    const float* bq = (const float*)d_in[4];
    const float* bk = (const float*)d_in[5];
    const float* bv = (const float*)d_in[6];
    const float* Wo = (const float*)d_in[7];
    const float* bo = (const float*)d_in[8];
    float* out = (float*)d_out;

    // Workspace carve-up (all offsets 256B-aligned).
    char* ws = (char*)d_ws;
    size_t off = 0;
    _Float16* xh  = (_Float16*)(ws + off); off += (size_t)SEQ * HIDDEN * 2;
    _Float16* Wqt = (_Float16*)(ws + off); off += (size_t)HEADS * HEAD_DIM * HIDDEN * 2;
    _Float16* Wkt = (_Float16*)(ws + off); off += (size_t)HEADS * HEAD_DIM * HIDDEN * 2;
    _Float16* Wvt = (_Float16*)(ws + off); off += (size_t)HEADS * HEAD_DIM * HIDDEN * 2;
    _Float16* Wot = (_Float16*)(ws + off); off += (size_t)HIDDEN * HIDDEN * 2;
    _Float16* Qh  = (_Float16*)(ws + off); off += (size_t)HEADS * SEQ * HEAD_DIM * 2;
    _Float16* Kh  = (_Float16*)(ws + off); off += (size_t)HEADS * SEQ * HEAD_DIM * 2;
    _Float16* Vt  = (_Float16*)(ws + off); off += (size_t)HEADS * HEAD_DIM * SEQ * 2;
    _Float16* ctx = (_Float16*)(ws + off); off += (size_t)SEQ * HIDDEN * 2;
    (void)ws_size; (void)n_in; (void)in_sizes; (void)out_size;

    // Stage 0: conversions / transposes
    {
        int n = SEQ * HIDDEN;                          // 3,145,728
        cvt_f32_f16<<<(n + 255) / 256, 256, 0, stream>>>(x, xh, n);
    }
    {
        int blocks = (HEADS * HEAD_DIM * HIDDEN) / 256; // 2304
        transpose_whead<<<blocks, 256, 0, stream>>>(Wq, Wqt);
        transpose_whead<<<blocks, 256, 0, stream>>>(Wk, Wkt);
        transpose_whead<<<blocks, 256, 0, stream>>>(Wv, Wvt);
        transpose_wo<<<(HIDDEN * HIDDEN) / 256, 256, 0, stream>>>(Wo, Wot);
    }

    // Stage 1: QKV projection — 3*12*128 wave tiles (32x64), 4 waves/block.
    qkv_kernel<<<(3 * HEADS * 128) / 4, 128, 0, stream>>>(
        xh, Wqt, Wkt, Wvt, bq, bk, bv, Qh, Kh, Vt);

    // Stage 2: flash attention
    attn_kernel<<<dim3(SEQ / 64, HEADS), 128, 0, stream>>>(Qh, Kh, Vt, ctx);

    // Stage 3: output projection — 12*128 wave tiles (32x64), 4 waves/block.
    proj_kernel<<<(HEADS * 128) / 4, 128, 0, stream>>>(ctx, Wot, bo, out);
}
